// LatentNeuralOperator_23175643529502
// MI455X (gfx1250) — matrix operand order repaired and checked by reading, not compiled, not run on hardware
//
#include <hip/hip_runtime.h>
#include <hip/hip_bf16.h>
#include <math.h>

// ---------------- problem constants (from reference) ----------------
#define LDd  32      // latent tokens
#define PDd  8       // phys dim
#define WW   64      // width
#define NLn  3       // layers
#define THh  64      // token hidden (LD*2)
#define CHh  128     // channel hidden (W*2)
#define TOKLD 68     // padded f32 row stride for tokens (bank-spread)
#define FRAG 512     // halves per swizzled 32x16 B-fragment tile (32 lanes x 16)

typedef __attribute__((ext_vector_type(16))) _Float16 v16h;
typedef __attribute__((ext_vector_type(8)))  _Float16 v8h;
typedef __attribute__((ext_vector_type(8)))  float    v8f;

// ---------------- LDS budget (halves/floats), shared by kernel + launcher ----------------
constexpr int H_wTi  = NLn * 4  * FRAG;   // tiW tiles: kt=1, nt=4
constexpr int H_wTo  = NLn * 4  * FRAG;   // toW tiles: kt=2, nt=2
constexpr int H_wCi  = NLn * 16 * FRAG;   // ciW tiles: kt=2, nt=8
constexpr int H_wCo  = NLn * 16 * FRAG;   // coW tiles: kt=4, nt=4
constexpr int H_pos  = LDd * WW;
constexpr int H_Aln  = 2048;              // max(64x32, 32x64)
constexpr int H_Hb   = 4096;              // max(64x64, 32x128)
constexpr int H_TOT  = H_wTi + H_wTo + H_wCi + H_wCo + H_pos + H_Aln + H_Hb;
constexpr int F_TOT  = LDd*TOKLD + PDd*WW
                     + NLn*THh + NLn*LDd + NLn*CHh + NLn*WW
                     + 4*NLn*WW + 2*WW + 2*WW + 2*WW + WW
                     + LDd + PDd + WW + WW + 4;
constexpr size_t SMEM_BYTES = (size_t)H_TOT * 2 + (size_t)F_TOT * 4;

// ---------------- helpers ----------------
__device__ __forceinline__ float gelu_exact(float x) {
  return 0.5f * x * (1.0f + erff(x * 0.7071067811865475f));
}

// A fragment: 16x32 f16, row-major activation with leading dim ld, base = &A[m0][k0].
// ISA 7.12.2: lanes 0-15 hold M=lane, Ks {0..7,16..23}; lanes 16-31 hold Ks {8..15,24..31}.
// Rows are 16B-aligned for all our ld choices -> two ds_load_b128.
__device__ __forceinline__ v16h frag_a(const _Float16* base, int ld, int lane) {
  const _Float16* r = base + (lane & 15) * ld + (lane >> 4) * 8;
  v8h lo = *(const v8h*)(r);
  v8h hi = *(const v8h*)(r + 16);
  return __builtin_shufflevector(lo, hi, 0,1,2,3,4,5,6,7,8,9,10,11,12,13,14,15);
}

// B fragment from pre-swizzled tile: each lane's 16 halves are contiguous.
__device__ __forceinline__ v16h frag_ld(const _Float16* tileBase, int lane) {
  const _Float16* r = tileBase + lane * 16;
  v8h lo = *(const v8h*)(r);
  v8h hi = *(const v8h*)(r + 8);
  return __builtin_shufflevector(lo, hi, 0,1,2,3,4,5,6,7,8,9,10,11,12,13,14,15);
}

__device__ __forceinline__ v8f wmma16(v16h a, v16h b, v8f c) {
  return __builtin_amdgcn_wmma_f32_16x16x32_f16(false, a, false, b, (short)0, c, false, false);
}

// Swizzle a row-major f32 weight [K][N] (ld = N) into B-fragment tiles.
// dst tile order: [kt*ntiles + nt], within tile: [lane][e];
// element (lane,e) <-> B[kt*32 + (lane>>4)*8 + (e&7) + (e>>3)*16][nt*16 + (lane&15)].
__device__ __forceinline__ void swizzleB(const float* src, _Float16* dst,
                                         int ld, int ktiles, int ntiles, int tid) {
  const int total = ktiles * ntiles * FRAG;
  for (int i = tid; i < total; i += 128) {
    int e = i & 15, lane = (i >> 4) & 31, tile = i >> 9;
    int nt = tile % ntiles, kt = tile / ntiles;
    int k = kt * 32 + (lane >> 4) * 8 + (e & 7) + (e >> 3) * 16;
    int n = nt * 16 + (lane & 15);
    dst[i] = (_Float16)src[k * ld + n];
  }
}

struct LNOParams {
  const float *phys, *latents, *pos_emb, *tokW, *tokb, *physW1, *physb1, *physW2, *physb2,
              *tn_g, *tn_b, *tiW, *tib, *toW, *tob, *cn_g, *cn_b, *ciW, *cib, *coW, *cob,
              *hn_g, *hn_b, *headW, *headb;
  float* out;
  int T;
};

// Persistent workgroup: one batch element, all T steps, state in LDS, WMMA matmuls.
__global__ __launch_bounds__(128, 1)
void lno_scan_kernel(LNOParams p) {
  extern __shared__ char smem[];
  _Float16* wTiF = (_Float16*)smem;                 // swizzled B-frags
  _Float16* wToF = wTiF + H_wTi;
  _Float16* wCiF = wToF + H_wTo;
  _Float16* wCoF = wCiF + H_wCi;
  _Float16* posH = wCoF + H_wCo;                    // [LD][W]
  _Float16* Aln  = posH + H_pos;                    // LN buffer (token: [W][LD], channel: [LD][W])
  _Float16* Hb   = Aln  + H_Aln;                    // hidden (H1: [64][64], H2: [32][128])
  float* fb    = (float*)(Hb + H_Hb);
  float* tokS  = fb;                  float* pW1   = tokS  + LDd*TOKLD;
  float* tibS  = pW1   + PDd*WW;      float* tobS  = tibS  + NLn*THh;
  float* cibS  = tobS  + NLn*LDd;     float* cobS  = cibS  + NLn*CHh;
  float* tngS  = cobS  + NLn*WW;      float* tnbS  = tngS  + NLn*WW;
  float* cngS  = tnbS  + NLn*WW;      float* cnbS  = cngS  + NLn*WW;
  float* hngS  = cnbS  + NLn*WW;      float* hnbS  = hngS  + WW;
  float* tokWs = hnbS  + WW;          float* tokbs = tokWs + WW;
  float* pb1S  = tokbs + WW;          float* pb2S  = pb1S  + WW;
  float* hWs   = pb2S  + WW;          float* curS  = hWs   + WW;
  float* physS = curS  + LDd;         float* hidS  = physS + PDd;
  float* pS    = hidS  + WW;          float* hbS   = pS    + WW;

  const int tid  = threadIdx.x;
  const int lane = tid & 31;
  const int wv   = tid >> 5;          // wave id 0..3
  const int b    = blockIdx.x;
  const int T    = p.T;

  // ---- one-time: weights -> f16 LDS in WMMA B-fragment layout ----
  for (int L = 0; L < NLn; ++L) {
    swizzleB(p.tiW + L*LDd*THh, wTiF + L*4*FRAG,  THh, 1, 4, tid);  // 32x64
    swizzleB(p.toW + L*THh*LDd, wToF + L*4*FRAG,  LDd, 2, 2, tid);  // 64x32
    swizzleB(p.ciW + L*WW*CHh,  wCiF + L*16*FRAG, CHh, 2, 8, tid);  // 64x128
    swizzleB(p.coW + L*CHh*WW,  wCoF + L*16*FRAG, WW,  4, 4, tid);  // 128x64
  }
  for (int i = tid; i < H_pos; i += 128) posH[i] = (_Float16)p.pos_emb[i];
  for (int i = tid; i < PDd*WW;  i += 128) pW1[i]  = p.physW1[i];
  for (int i = tid; i < NLn*THh; i += 128) tibS[i] = p.tib[i];
  for (int i = tid; i < NLn*LDd; i += 128) tobS[i] = p.tob[i];
  for (int i = tid; i < NLn*CHh; i += 128) cibS[i] = p.cib[i];
  for (int i = tid; i < NLn*WW;  i += 128) { cobS[i] = p.cob[i]; tngS[i] = p.tn_g[i];
    tnbS[i] = p.tn_b[i]; cngS[i] = p.cn_g[i]; cnbS[i] = p.cn_b[i]; }
  if (tid < WW) { hngS[tid] = p.hn_g[tid]; hnbS[tid] = p.hn_b[tid];
    tokWs[tid] = p.tokW[tid]; tokbs[tid] = p.tokb[tid];
    pb1S[tid] = p.physb1[tid]; pb2S[tid] = p.physb2[tid]; hWs[tid] = p.headW[tid]; }
  if (tid < LDd) curS[tid] = p.latents[(size_t)b * LDd + tid];
  if (tid == 0) hbS[0] = p.headb[0];
  __syncthreads();

  for (int t = 0; t < T; ++t) {
    // -------- inputs for this step --------
    if (tid < PDd) physS[tid] = p.phys[((size_t)b * T + t) * PDd + tid];
    if (tid == 0 && t + 1 < T)
      __builtin_prefetch(&p.phys[((size_t)b * T + t + 1) * PDd], 0, 0); // global_prefetch_b8
    __syncthreads();

    // -------- phys MLP (tiny: scalar VALU) --------
    if (tid < WW) {
      float h = pb1S[tid];
#pragma unroll
      for (int k = 0; k < PDd; ++k) h += physS[k] * pW1[k * WW + tid];
      hidS[tid] = gelu_exact(h);
    }
    __syncthreads();
    if (tid < WW) {
      float acc = pb2S[tid];
      for (int k = 0; k < WW; ++k) acc += hidS[k] * p.physW2[k * WW + tid]; // L2-resident 16KB
      pS[tid] = acc;
    }
    __syncthreads();

    // -------- token embedding: tok[l][w] = cur[l]*tokW[w]+tokb[w]+p[w]+pos[l][w] --------
#pragma unroll
    for (int i = 0; i < (LDd * WW) / 128; ++i) {
      int e = tid + i * 128, l = e >> 6, w = e & 63;
      tokS[l * TOKLD + w] = curS[l] * tokWs[w] + tokbs[w] + pS[w] + (float)posH[e];
    }
    __syncthreads();

    for (int L = 0; L < NLn; ++L) {
      // ===== token-mix LN (over W), write TRANSPOSED f16: Aln[w*32 + l] =====
      {
        int l = tid >> 2, sub = tid & 3, wb = sub * 16;
        float x[16], s = 0.f, s2 = 0.f;
#pragma unroll
        for (int j = 0; j < 16; ++j) { float v = tokS[l * TOKLD + wb + j]; x[j] = v; s += v; s2 += v * v; }
        s  += __shfl_xor(s, 1);  s  += __shfl_xor(s, 2);
        s2 += __shfl_xor(s2, 1); s2 += __shfl_xor(s2, 2);
        float mean = s * (1.0f / WW);
        float rinv = rsqrtf(s2 * (1.0f / WW) - mean * mean + 1e-5f);
#pragma unroll
        for (int j = 0; j < 16; ++j) {
          int w = wb + j;
          Aln[w * LDd + l] = (_Float16)((x[j] - mean) * rinv * tngS[L * WW + w] + tnbS[L * WW + w]);
        }
      }
      __syncthreads();

      // ===== mm1: H1(64x64) = gelu(Aln(64x32) @ tiW_L(32x64) + tib) ; M-tile = wave =====
      {
        v16h a = frag_a(&Aln[(wv * 16) * LDd], LDd, lane);
#pragma unroll
        for (int nt = 0; nt < 4; ++nt) {
          v16h bf = frag_ld(wTiF + (L * 4 + nt) * FRAG, lane);
          v8f acc = {};
          acc = wmma16(a, bf, acc);
          int n = nt * 16 + (lane & 15), mo = (lane >> 4) * 8;
          float bias = tibS[L * THh + n];
#pragma unroll
          for (int r = 0; r < 8; ++r)
            Hb[(wv * 16 + mo + r) * THh + n] = (_Float16)gelu_exact(acc[r] + bias);
        }
      }
      // M-stationary: this wave's mm2 A-rows were produced by this wave (LDS in-order) -> no barrier

      // ===== mm2: tok += transpose(H1(64x64) @ toW_L(64x32) + tob) =====
      {
        v16h a0 = frag_a(&Hb[(wv * 16) * THh + 0],  THh, lane);
        v16h a1 = frag_a(&Hb[(wv * 16) * THh + 32], THh, lane);
#pragma unroll
        for (int nt = 0; nt < 2; ++nt) {
          v16h b0 = frag_ld(wToF + (L * 4 + 0 * 2 + nt) * FRAG, lane);
          v16h b1 = frag_ld(wToF + (L * 4 + 1 * 2 + nt) * FRAG, lane);
          v8f acc = {};
          acc = wmma16(a0, b0, acc);
          acc = wmma16(a1, b1, acc);
          int n = nt * 16 + (lane & 15), mo = (lane >> 4) * 8;   // n = latent l, m = width w
          float bias = tobS[L * LDd + n];
#pragma unroll
          for (int r = 0; r < 8; ++r)
            tokS[n * TOKLD + (wv * 16 + mo + r)] += acc[r] + bias;
        }
      }
      __syncthreads();

      // ===== channel LN (over W), row-major f16: Aln[l*64 + w] =====
      {
        int l = tid >> 2, sub = tid & 3, wb = sub * 16;
        float x[16], s = 0.f, s2 = 0.f;
#pragma unroll
        for (int j = 0; j < 16; ++j) { float v = tokS[l * TOKLD + wb + j]; x[j] = v; s += v; s2 += v * v; }
        s  += __shfl_xor(s, 1);  s  += __shfl_xor(s, 2);
        s2 += __shfl_xor(s2, 1); s2 += __shfl_xor(s2, 2);
        float mean = s * (1.0f / WW);
        float rinv = rsqrtf(s2 * (1.0f / WW) - mean * mean + 1e-5f);
#pragma unroll
        for (int j = 0; j < 16; ++j) {
          int w = wb + j;
          Aln[l * WW + w] = (_Float16)((x[j] - mean) * rinv * cngS[L * WW + w] + cnbS[L * WW + w]);
        }
      }
      __syncthreads();

      // ===== mm3: H2(32x128) = gelu(Zn(32x64) @ ciW_L(64x128) + cib) ; N-tiles {2wv,2wv+1} =====
      {
        v16h a00 = frag_a(&Aln[0 * WW + 0],   WW, lane);
        v16h a01 = frag_a(&Aln[0 * WW + 32],  WW, lane);
        v16h a10 = frag_a(&Aln[16 * WW + 0],  WW, lane);
        v16h a11 = frag_a(&Aln[16 * WW + 32], WW, lane);
#pragma unroll
        for (int q = 0; q < 2; ++q) {
          int nt = wv * 2 + q;
          v16h b0 = frag_ld(wCiF + (L * 16 + 0 * 8 + nt) * FRAG, lane);
          v16h b1 = frag_ld(wCiF + (L * 16 + 1 * 8 + nt) * FRAG, lane);
          int n = nt * 16 + (lane & 15), mo = (lane >> 4) * 8;
          float bias = cibS[L * CHh + n];
          v8f acc0 = {};
          acc0 = wmma16(a00, b0, acc0);
          acc0 = wmma16(a01, b1, acc0);
#pragma unroll
          for (int r = 0; r < 8; ++r)
            Hb[(mo + r) * CHh + n] = (_Float16)gelu_exact(acc0[r] + bias);
          v8f acc1 = {};
          acc1 = wmma16(a10, b0, acc1);
          acc1 = wmma16(a11, b1, acc1);
#pragma unroll
          for (int r = 0; r < 8; ++r)
            Hb[(16 + mo + r) * CHh + n] = (_Float16)gelu_exact(acc1[r] + bias);
        }
      }
      __syncthreads();  // mm4 reads H2 columns produced by all waves

      // ===== mm4: tok += H2(32x128) @ coW_L(128x64) + cob ; N-tile = wave =====
      {
        v16h bf[4];
#pragma unroll
        for (int kk = 0; kk < 4; ++kk)
          bf[kk] = frag_ld(wCoF + (L * 16 + kk * 4 + wv) * FRAG, lane);
#pragma unroll
        for (int mt = 0; mt < 2; ++mt) {
          v8f acc = {};
#pragma unroll
          for (int kk = 0; kk < 4; ++kk) {
            v16h a = frag_a(&Hb[(mt * 16) * CHh + kk * 32], CHh, lane);
            acc = wmma16(a, bf[kk], acc);
          }
          int n = wv * 16 + (lane & 15), mo = (lane >> 4) * 8;
          float bias = cobS[L * WW + n];
#pragma unroll
          for (int r = 0; r < 8; ++r)
            tokS[(mt * 16 + mo + r) * TOKLD + n] += acc[r] + bias;
        }
      }
      __syncthreads();
    } // layers

    // -------- head: delta = LN(tok) @ headW + headb; cur = clip(cur+delta) --------
    {
      int l = tid >> 2, sub = tid & 3, wb = sub * 16;
      float x[16], s = 0.f, s2 = 0.f;
#pragma unroll
      for (int j = 0; j < 16; ++j) { float v = tokS[l * TOKLD + wb + j]; x[j] = v; s += v; s2 += v * v; }
      s  += __shfl_xor(s, 1);  s  += __shfl_xor(s, 2);
      s2 += __shfl_xor(s2, 1); s2 += __shfl_xor(s2, 2);
      float mean = s * (1.0f / WW);
      float rinv = rsqrtf(s2 * (1.0f / WW) - mean * mean + 1e-5f);
      float part = 0.f;
#pragma unroll
      for (int j = 0; j < 16; ++j) {
        int w = wb + j;
        part += ((x[j] - mean) * rinv * hngS[w] + hnbS[w]) * hWs[w];
      }
      part += __shfl_xor(part, 1); part += __shfl_xor(part, 2);
      if (sub == 0) {
        float nc = curS[l] + part + hbS[0];
        nc = fminf(fmaxf(nc, 0.0f), 1.0f);
        curS[l] = nc;
        p.out[((size_t)b * T + t) * LDd + l] = nc;
      }
    }
    __syncthreads();
  } // t
}

extern "C" void kernel_launch(void* const* d_in, const int* in_sizes, int n_in,
                              void* d_out, int out_size, void* d_ws, size_t ws_size,
                              hipStream_t stream) {
  (void)n_in; (void)d_ws; (void)ws_size; (void)out_size;
  LNOParams P;
  P.phys   = (const float*)d_in[0];  P.latents = (const float*)d_in[1];
  P.pos_emb= (const float*)d_in[2];  P.tokW    = (const float*)d_in[3];
  P.tokb   = (const float*)d_in[4];  P.physW1  = (const float*)d_in[5];
  P.physb1 = (const float*)d_in[6];  P.physW2  = (const float*)d_in[7];
  P.physb2 = (const float*)d_in[8];  P.tn_g    = (const float*)d_in[9];
  P.tn_b   = (const float*)d_in[10]; P.tiW     = (const float*)d_in[11];
  P.tib    = (const float*)d_in[12]; P.toW     = (const float*)d_in[13];
  P.tob    = (const float*)d_in[14]; P.cn_g    = (const float*)d_in[15];
  P.cn_b   = (const float*)d_in[16]; P.ciW     = (const float*)d_in[17];
  P.cib    = (const float*)d_in[18]; P.coW     = (const float*)d_in[19];
  P.cob    = (const float*)d_in[20]; P.hn_g    = (const float*)d_in[21];
  P.hn_b   = (const float*)d_in[22]; P.headW   = (const float*)d_in[23];
  P.headb  = (const float*)d_in[24];
  P.out    = (float*)d_out;
  const int Bn = in_sizes[1] / LDd;                 // latents is [B, LD]
  P.T = in_sizes[0] / (Bn * PDd);                   // phys is [B, T, PD]
  lno_scan_kernel<<<dim3(Bn), dim3(128), SMEM_BYTES, stream>>>(P);
}
